// MultiheadAttnetion_68642167325178
// MI455X (gfx1250) — matrix operand order repaired and checked by reading, not compiled, not run on hardware
//
#include <hip/hip_runtime.h>
#include <hip/hip_bf16.h>

// ---------- types ----------
typedef __attribute__((ext_vector_type(16))) __bf16    v16bf;
typedef __attribute__((ext_vector_type(8)))  float     v8f;
typedef __attribute__((ext_vector_type(4)))  unsigned int   u32x4;
typedef __attribute__((ext_vector_type(4)))  float          f32x4;
typedef __attribute__((ext_vector_type(4)))  unsigned short u16x4;

typedef unsigned short u16;
typedef unsigned int   u32;

union BFrag { v16bf v; u32x4 q[2]; };

__device__ __forceinline__ u16 f2bf(float f) {
    u32 u = __float_as_uint(f);
    u32 r = (u + 0x7FFFu + ((u >> 16) & 1u)) >> 16;   // round-to-nearest-even
    return (u16)r;
}

__device__ __forceinline__ u32x4 ldv4(const u16* p) { return *(const u32x4*)p; }
__device__ __forceinline__ void  stv4(u16* p, u32x4 v) { *(u32x4*)p = v; }

// LDS byte offset = low 32 bits of the generic (flat-aperture) address.
__device__ __forceinline__ u32 lds_off(const void* p) {
    return (u32)(unsigned long long)p;
}

// Async DMA: global -> LDS, 16B per lane, tracked by ASYNCcnt.
__device__ __forceinline__ void async_ld_b128(u32 ldsoff, const void* gp) {
    asm volatile("global_load_async_to_lds_b128 %0, %1, off"
                 :: "v"(ldsoff), "v"((unsigned long long)gp) : "memory");
}
__device__ __forceinline__ void wait_async0() {
    asm volatile("s_wait_asynccnt 0x0" ::: "memory");
}

// xor-shuffle within each 16-lane row on the VALU (v_permlane16_b32),
// keeping reductions off the LDS pipe. Selector nibble for lane i is i^mask.
__device__ __forceinline__ float pl16(float x, u32 s0, u32 s1) {
    u32 u = __float_as_uint(x);
    return __uint_as_float(__builtin_amdgcn_permlane16(u, u, s0, s1, false, false));
}
__device__ __forceinline__ float redmax16(float x) {
    x = fmaxf(x, pl16(x, 0x67452301u, 0xEFCDAB89u));  // mask 1
    x = fmaxf(x, pl16(x, 0x54761032u, 0xDCFE98BAu));  // mask 2
    x = fmaxf(x, pl16(x, 0x32107654u, 0xBA98FEDCu));  // mask 4
    x = fmaxf(x, pl16(x, 0xFEDCBA98u, 0x76543210u));  // mask 8
    return x;
}
__device__ __forceinline__ float redsum16(float x) {
    x += pl16(x, 0x67452301u, 0xEFCDAB89u);
    x += pl16(x, 0x54761032u, 0xDCFE98BAu);
    x += pl16(x, 0x32107654u, 0xBA98FEDCu);
    x += pl16(x, 0xFEDCBA98u, 0x76543210u);
    return x;
}

// ---------- f32 -> bf16 conversion ----------
__global__ void cvt_f32_to_bf16_kernel(const f32x4* __restrict__ src,
                                       u16x4* __restrict__ dst, int n4) {
    for (int i = blockIdx.x * blockDim.x + threadIdx.x; i < n4;
         i += gridDim.x * blockDim.x) {
        f32x4 f = src[i];
        u16x4 o;
        o.x = f2bf(f.x); o.y = f2bf(f.y); o.z = f2bf(f.z); o.w = f2bf(f.w);
        dst[i] = o;
    }
}

// ---------- tiled bf16 WMMA GEMM:  out[M,N] = A[M,K] @ W[K,N] + bias ----------
// Block tile 128x128, BK=32, 8 waves (4 along M x 2 along N), wave tile 32x64.
// Double-buffered: A tile via async global->LDS DMA, W tile via VGPR transpose.
template<bool OUT_BF16>
__global__ __launch_bounds__(256)
void gemm_bf16_kernel(const u16* __restrict__ A, const u16* __restrict__ W,
                      const float* __restrict__ bias, void* __restrict__ out,
                      int M, int N, int K)
{
    constexpr int BM = 128, BN = 128, BK = 32;
    constexpr int ASTR = 48, BSTR = 48;            // row strides (elems), 16B-aligned rows
    __shared__ u16 lsA[2][BM * ASTR];              // [row][k]  row-major
    __shared__ u16 lsB[2][BN * BSTR];              // [n][k]    (W tile transposed)

    const int t    = threadIdx.x;
    const int lane = t & 31;
    const int wave = t >> 5;
    const int wm   = wave >> 1;                    // 0..3 -> rows wm*32
    const int wn   = wave & 1;                     // 0..1 -> cols wn*64
    const int nl   = lane & 15;
    const int half = lane >> 4;
    const int m0   = blockIdx.y * BM;
    const int n0   = blockIdx.x * BN;

    auto stageA = [&](int k0, int buf) {
        #pragma unroll
        for (int i = 0; i < 2; ++i) {
            int c = t + i * 256;
            int row = c >> 2, cc = c & 3;
            async_ld_b128(lds_off(&lsA[buf][row * ASTR + cc * 8]),
                          &A[(size_t)(m0 + row) * K + k0 + cc * 8]);
        }
    };
    auto stageB = [&](int k0, int buf) {
        #pragma unroll
        for (int i = 0; i < 2; ++i) {
            int c = t + i * 256;
            int kr = c >> 4, cc = c & 15;
            u32x4 d = ldv4(&W[(size_t)(k0 + kr) * N + n0 + cc * 8]);
            const u16* dp = (const u16*)&d;
            #pragma unroll
            for (int j = 0; j < 8; ++j)
                lsB[buf][(cc * 8 + j) * BSTR + kr] = dp[j];
        }
    };

    v8f acc[2][4];
    #pragma unroll
    for (int i = 0; i < 2; ++i)
        #pragma unroll
        for (int j = 0; j < 4; ++j)
            acc[i][j] = v8f{0.f,0.f,0.f,0.f,0.f,0.f,0.f,0.f};

    stageA(0, 0);
    stageB(0, 0);
    wait_async0();
    __syncthreads();

    const int NT = K / BK;
    for (int it = 0; it < NT; ++it) {
        const int buf = it & 1;
        if (it + 1 < NT) {                         // overlap next-tile DMA with WMMAs
            stageA((it + 1) * BK, buf ^ 1);
            stageB((it + 1) * BK, buf ^ 1);
        }

        // ---- A fragments (16x32 bf16; lanes=M, K split by lane-half) ----
        BFrag a[2];
        #pragma unroll
        for (int mf = 0; mf < 2; ++mf) {
            int row = wm * 32 + mf * 16 + nl;
            a[mf].q[0] = ldv4(&lsA[buf][row * ASTR +      8 * half]);
            a[mf].q[1] = ldv4(&lsA[buf][row * ASTR + 16 + 8 * half]);
        }
        // ---- B fragments (32x16; lanes=N, K split by lane-half) + WMMAs ----
        #pragma unroll
        for (int f = 0; f < 4; ++f) {
            BFrag b;
            int n = wn * 64 + f * 16 + nl;
            b.q[0] = ldv4(&lsB[buf][n * BSTR +      8 * half]);
            b.q[1] = ldv4(&lsB[buf][n * BSTR + 16 + 8 * half]);
            #pragma unroll
            for (int mf = 0; mf < 2; ++mf)
                acc[mf][f] = __builtin_amdgcn_wmma_f32_16x16x32_bf16(
                    false, a[mf].v, false, b.v, (short)0, acc[mf][f], false, false);
        }

        wait_async0();                             // next tile fully in LDS
        __syncthreads();                           // everyone done reading cur tile
    }

    // ---- epilogue: bias + store ----
    #pragma unroll
    for (int f = 0; f < 4; ++f) {
        int col = n0 + wn * 64 + f * 16 + nl;
        float bv = bias[col];
        #pragma unroll
        for (int mf = 0; mf < 2; ++mf) {
            #pragma unroll
            for (int r = 0; r < 8; ++r) {
                int row = m0 + wm * 32 + mf * 16 + r + 8 * half;
                float v = acc[mf][f][r] + bv;
                if (OUT_BF16) ((u16*)out)[(size_t)row * N + col] = f2bf(v);
                else          ((float*)out)[(size_t)row * N + col] = v;
            }
        }
    }
}

// ---------- fused flash attention over contiguous [S,64] head blocks ----------
// grid = (S/64, B*H), block = 128 (4 waves); each wave owns 16 query rows.
__global__ __launch_bounds__(128)
void flash_attn_kernel(const u16* __restrict__ Qb, const u16* __restrict__ Kb,
                       const u16* __restrict__ Vb, u16* __restrict__ Ob)
{
    constexpr int S = 2048, DK = 64, KB = 32;
    constexpr int KSTR = 72, VSTR = 48, PSTR = 48, QSTR = 72;
    __shared__ u16 lsK[KB * KSTR];        // [key][dk]   (already B^T layout for QK^T)
    __shared__ u16 lsV[DK * VSTR];        // [dk][key]   (V transposed for P@V)
    __shared__ u16 lsQ[64 * QSTR];        // [qrow][dk]
    __shared__ u16 lsP[4][16 * PSTR];     // per-wave probs [qrow16][key32]

    const int t    = threadIdx.x;
    const int lane = t & 31, wave = t >> 5;
    const int nl   = lane & 15, half = lane >> 4;
    const int bh   = blockIdx.y;
    const int q0   = blockIdx.x * 64;
    const size_t base = (size_t)bh * S * DK;
    const u16* Q  = Qb + base;
    const u16* Kp = Kb + base;
    const u16* Vp = Vb + base;

    // stage Q tile 64x64 (async DMA)
    #pragma unroll
    for (int i = 0; i < 4; ++i) {
        int c = t + i * 128;
        int row = c >> 3, cc = c & 7;
        async_ld_b128(lds_off(&lsQ[row * QSTR + cc * 8]),
                      &Q[(size_t)(q0 + row) * DK + cc * 8]);
    }
    wait_async0();
    __syncthreads();

    // per-wave Q A-fragments: dk contraction split into 2 steps of 32
    BFrag aQ[2];
    int qrow = wave * 16 + nl;
    #pragma unroll
    for (int st = 0; st < 2; ++st) {
        aQ[st].q[0] = ldv4(&lsQ[qrow * QSTR + st * 32 + 16 * half]);
        aQ[st].q[1] = ldv4(&lsQ[qrow * QSTR + st * 32 + 16 * half + 8]);
    }

    v8f O[4];
    #pragma unroll
    for (int f = 0; f < 4; ++f) O[f] = v8f{0.f,0.f,0.f,0.f,0.f,0.f,0.f,0.f};
    float mrun[8], lrun[8];
    #pragma unroll
    for (int r = 0; r < 8; ++r) { mrun[r] = -3.0e38f; lrun[r] = 0.f; }

    for (int kb = 0; kb < S / KB; ++kb) {
        __syncthreads();
        // stage K tile [32][64] row-major via async DMA
        #pragma unroll
        for (int i = 0; i < 2; ++i) {
            int c = t + i * 128;
            int row = c >> 3, cc = c & 7;
            async_ld_b128(lds_off(&lsK[row * KSTR + cc * 8]),
                          &Kp[(size_t)(kb * KB + row) * DK + cc * 8]);
        }
        // stage V tile transposed -> lsV[dk][key] (needs VGPR transpose)
        #pragma unroll
        for (int i = 0; i < 2; ++i) {
            int c = t + i * 128;
            int row = c >> 3, cc = c & 7;
            u32x4 d = ldv4(&Vp[(size_t)(kb * KB + row) * DK + cc * 8]);
            const u16* dp = (const u16*)&d;
            #pragma unroll
            for (int j = 0; j < 8; ++j)
                lsV[(cc * 8 + j) * VSTR + row] = dp[j];
        }
        wait_async0();
        __syncthreads();

        // ---- logits = Q @ K^T (two 16-col fragments over 32 keys) ----
        v8f c0 = v8f{0.f,0.f,0.f,0.f,0.f,0.f,0.f,0.f};
        v8f c1 = c0;
        #pragma unroll
        for (int st = 0; st < 2; ++st) {
            BFrag b0, b1;
            b0.q[0] = ldv4(&lsK[nl * KSTR + st * 32 + 16 * half]);
            b0.q[1] = ldv4(&lsK[nl * KSTR + st * 32 + 16 * half + 8]);
            b1.q[0] = ldv4(&lsK[(16 + nl) * KSTR + st * 32 + 16 * half]);
            b1.q[1] = ldv4(&lsK[(16 + nl) * KSTR + st * 32 + 16 * half + 8]);
            c0 = __builtin_amdgcn_wmma_f32_16x16x32_bf16(false, aQ[st].v, false, b0.v,
                                                         (short)0, c0, false, false);
            c1 = __builtin_amdgcn_wmma_f32_16x16x32_bf16(false, aQ[st].v, false, b1.v,
                                                         (short)0, c1, false, false);
        }

        // ---- online softmax; reductions on VALU via v_permlane16_b32 ----
        #pragma unroll
        for (int r = 0; r < 8; ++r) {
            float l0 = c0[r] * 0.125f, l1 = c1[r] * 0.125f;   // 1/sqrt(64)
            float bm = redmax16(fmaxf(l0, l1));
            float nm = fmaxf(mrun[r], bm);
            float sc = __expf(mrun[r] - nm);
            mrun[r] = nm;
            float p0 = __expf(l0 - nm), p1 = __expf(l1 - nm);
            float rs = redsum16(p0 + p1);
            lrun[r] = lrun[r] * sc + rs;
            #pragma unroll
            for (int f = 0; f < 4; ++f) O[f][r] *= sc;
            int prow = r + 8 * half;                      // C-layout row -> LDS row
            lsP[wave][prow * PSTR + nl]      = f2bf(p0);
            lsP[wave][prow * PSTR + 16 + nl] = f2bf(p1);
        }
        asm volatile("s_wait_dscnt 0x0" ::: "memory");    // P store -> A-frag load

        // ---- O += P @ V ----
        BFrag aP;
        aP.q[0] = ldv4(&lsP[wave][nl * PSTR + 16 * half]);
        aP.q[1] = ldv4(&lsP[wave][nl * PSTR + 16 * half + 8]);
        #pragma unroll
        for (int f = 0; f < 4; ++f) {
            BFrag bV;
            int c = f * 16 + nl;
            bV.q[0] = ldv4(&lsV[c * VSTR + 16 * half]);
            bV.q[1] = ldv4(&lsV[c * VSTR + 16 * half + 8]);
            O[f] = __builtin_amdgcn_wmma_f32_16x16x32_bf16(false, aP.v, false, bV.v,
                                                           (short)0, O[f], false, false);
        }
    }

    // ---- normalize + store bf16 (layout already matches reshape-back) ----
    #pragma unroll
    for (int r = 0; r < 8; ++r) {
        float inv = 1.0f / lrun[r];
        int row = q0 + wave * 16 + r + 8 * half;
        #pragma unroll
        for (int f = 0; f < 4; ++f)
            Ob[base + (size_t)row * DK + f * 16 + nl] = f2bf(O[f][r] * inv);
    }
}

// ---------- host launch ----------
extern "C" void kernel_launch(void* const* d_in, const int* in_sizes, int n_in,
                              void* d_out, int out_size, void* d_ws, size_t ws_size,
                              hipStream_t stream) {
    constexpr int B = 4, S = 2048, E = 1024, H = 16;
    const int M = B * S;                        // 8192
    const float* x  = (const float*)d_in[0];
    const float* Wq = (const float*)d_in[1];
    const float* bq = (const float*)d_in[2];
    const float* Wk = (const float*)d_in[3];
    const float* bk = (const float*)d_in[4];
    const float* Wv = (const float*)d_in[5];
    const float* bv = (const float*)d_in[6];
    const float* Wo = (const float*)d_in[7];
    const float* bo = (const float*)d_in[8];

    char* ws = (char*)d_ws;
    const size_t xeb = (size_t)M * E * 2;       // 16 MB
    const size_t web = (size_t)E * E * 2;       //  2 MB
    u16* xb  = (u16*)ws;            ws += xeb;
    u16* Wqb = (u16*)ws;            ws += web;
    u16* Wkb = (u16*)ws;            ws += web;
    u16* Wvb = (u16*)ws;            ws += web;
    u16* Wob = (u16*)ws;            ws += web;
    u16* Qb  = (u16*)ws;            ws += xeb;
    u16* Kb  = (u16*)ws;            ws += xeb;
    u16* Vb  = (u16*)ws;            ws += xeb;
    u16* Sb  = (u16*)ws;            ws += xeb;

    // f32 -> bf16
    cvt_f32_to_bf16_kernel<<<2048, 256, 0, stream>>>((const f32x4*)x,  (u16x4*)xb,  M * E / 4);
    cvt_f32_to_bf16_kernel<<<1024, 256, 0, stream>>>((const f32x4*)Wq, (u16x4*)Wqb, E * E / 4);
    cvt_f32_to_bf16_kernel<<<1024, 256, 0, stream>>>((const f32x4*)Wk, (u16x4*)Wkb, E * E / 4);
    cvt_f32_to_bf16_kernel<<<1024, 256, 0, stream>>>((const f32x4*)Wv, (u16x4*)Wvb, E * E / 4);
    cvt_f32_to_bf16_kernel<<<1024, 256, 0, stream>>>((const f32x4*)Wo, (u16x4*)Wob, E * E / 4);

    // QKV projections (bf16 out + bias)
    dim3 gg(E / 128, M / 128);                  // (8, 64)
    gemm_bf16_kernel<true><<<gg, 256, 0, stream>>>(xb, Wqb, bq, Qb, M, E, E);
    gemm_bf16_kernel<true><<<gg, 256, 0, stream>>>(xb, Wkb, bk, Kb, M, E, E);
    gemm_bf16_kernel<true><<<gg, 256, 0, stream>>>(xb, Wvb, bv, Vb, M, E, E);

    // fused attention per (b,h) contiguous [S,64] block
    flash_attn_kernel<<<dim3(S / 64, B * H), 128, 0, stream>>>(Qb, Kb, Vb, Sb);

    // output projection (f32 out + bias) -> d_out
    gemm_bf16_kernel<false><<<gg, 256, 0, stream>>>(Sb, Wob, bo, d_out, M, E, E);
}
